// WindowAttention_20366734917788
// MI455X (gfx1250) — matrix operand order repaired and checked by reading, not compiled, not run on hardware
//
#include <hip/hip_runtime.h>
#include <hip/hip_bf16.h>

typedef __attribute__((ext_vector_type(16))) __bf16 bf16x16;
typedef __attribute__((ext_vector_type(8)))  float  f32x8;
typedef __attribute__((ext_vector_type(4)))  int    v4i;
typedef unsigned short u16;

#define N_TOK 343
#define NH 6
#define HD 32
#define CDIM 192
#define NW 32

#define AS1 __attribute__((address_space(1)))
#define AS3 __attribute__((address_space(3)))

#if __has_builtin(__builtin_amdgcn_global_load_async_to_lds_b128)
#define HAVE_ASYNC_LDS 1
#else
#define HAVE_ASYNC_LDS 0
#endif

__device__ __forceinline__ u16 f2bf(float f) {
  unsigned u = __builtin_bit_cast(unsigned, f);
  u += 0x7fffu + ((u >> 16) & 1u);          // round-to-nearest-even
  return (u16)(u >> 16);
}
__device__ __forceinline__ __bf16 bfld(u16 v) {
  return __builtin_bit_cast(__bf16, v);
}

#if HAVE_ASYNC_LDS
__device__ __forceinline__ void async_ld_b128(const u16* g, u16* l) {
  // b128 builtin takes int4* operands: global src in AS1, LDS dst in AS3.
  // Generic LDS addr: low 32 bits == LDS offset (ISA flat->LDS truncation).
  __builtin_amdgcn_global_load_async_to_lds_b128(
      (AS1 v4i*)(unsigned long long)(const void*)g,
      (AS3 v4i*)(unsigned)(unsigned long long)(void*)l, 0, 0);
}
__device__ __forceinline__ void wait_async0() {
#if __has_builtin(__builtin_amdgcn_s_wait_asynccnt)
  __builtin_amdgcn_s_wait_asynccnt(0);
#else
  asm volatile("s_wait_asynccnt 0x0" ::: "memory");
#endif
}
#endif

// ---------------------------------------------------------------- convert
__global__ void __launch_bounds__(256) cvt_bf16(const float* __restrict__ s,
                                                u16* __restrict__ d, int n) {
  int i = blockIdx.x * 256 + threadIdx.x;
  if (i < n) d[i] = f2bf(s[i]);
}

// ---------------------------------------------------------------- QKV GEMM
// out(43904 x 576) = xb(43904 x 192) @ qkv_w^T, scatter into q/k/v (B,H,N,32) bf16.
__global__ void __launch_bounds__(256) qkv_gemm(
    const u16* __restrict__ xb, const u16* __restrict__ wb,
    const float* __restrict__ qkv_b,
    u16* __restrict__ qo, u16* __restrict__ ko, u16* __restrict__ vo) {
  const int wave = threadIdx.x >> 5, lane = threadIdx.x & 31;
  const int lo = lane & 15, hi = lane >> 4;
  const int tile = blockIdx.x * 8 + wave;           // 98784 tiles total
  const int tm = tile / 36, tn = tile % 36;
  const u16* arow = xb + (long)(tm * 16 + lo) * CDIM;
  const u16* brow = wb + (long)(tn * 16 + lo) * CDIM;
  __builtin_prefetch(brow, 0, 3);
  f32x8 acc = {};
#pragma unroll
  for (int ks = 0; ks < 6; ++ks) {
    const int k0 = ks * 32;
    bf16x16 a, b;
#pragma unroll
    for (int i = 0; i < 8; ++i) {
      a[i]     = bfld(arow[k0 + hi * 8 + i]);       // A 16x32 lane layout
      a[i + 8] = bfld(arow[k0 + 16 + hi * 8 + i]);
    }
#pragma unroll
    for (int i = 0; i < 16; ++i) b[i] = bfld(brow[k0 + (i & 8) * 2 + hi * 8 + (i & 7)]);  // B 32x16
    acc = __builtin_amdgcn_wmma_f32_16x16x32_bf16(false, a, false, b,
                                                  (short)0, acc, false, false);
  }
  const int jc = tn * 16 + lo;
  const int part = jc / CDIM, within = jc % CDIM;
  const int head = within >> 5, d = within & 31;
  u16* dst = (part == 0) ? qo : ((part == 1) ? ko : vo);
  const float bj  = qkv_b[jc];
  const float scl = (part == 0) ? 0.17677669529663687f : 1.0f;  // hd^-0.5 on q
#pragma unroll
  for (int v = 0; v < 8; ++v) {
    int r  = tm * 16 + hi * 8 + v;                  // global row (C layout)
    int bw = r / N_TOK, tok = r - bw * N_TOK;
    dst[((long)(bw * NH + head) * N_TOK + tok) * HD + d] = f2bf((acc[v] + bj) * scl);
  }
}

// ---------------------------------------------------------------- fused attention
// grid (B*H, 3); each wave owns one 16-row tile; flash softmax over 11 chunks of 32.
// K/V chunks staged with GLOBAL_LOAD_ASYNC_TO_LDS_B128 (ASYNCcnt) when available.
__global__ void __launch_bounds__(256) win_attn(
    const u16* __restrict__ qb, const u16* __restrict__ kb, const u16* __restrict__ vb,
    const float* __restrict__ mask, const float* __restrict__ rpb,
    const int* __restrict__ relidx, u16* __restrict__ aob) {
  __shared__ u16 Ks[32 * 32];        // [tok][d]
  __shared__ u16 Vs[32 * 32];        // [tok][d] (untransposed; strided reads)
  __shared__ u16 Pst[8 * 16 * 32];   // per-wave P transpose staging
  const int wave = threadIdx.x >> 5, lane = threadIdx.x & 31;
  const int lo = lane & 15, hi = lane >> 4;
  const int bh = blockIdx.x;
  const int b = bh / NH, h = bh % NH;
  const int wmsk = b & (NW - 1);                     // b % nW
  const int rowBase = (blockIdx.y * 8 + wave) * 16;
  const long base = (long)bh * N_TOK * HD;

  // q A-fragment (16x32, K = hd), rows >= N zeroed
  bf16x16 qa;
  {
    int mg = rowBase + lo;
    bool ok = mg < N_TOK;
    const u16* qrow = qb + base + (long)(ok ? mg : 0) * HD;
#pragma unroll
    for (int i = 0; i < 8; ++i) {
      qa[i]     = ok ? bfld(qrow[hi * 16 + i])      : bfld((u16)0);
      qa[i + 8] = ok ? bfld(qrow[hi * 16 + 8 + i]) : bfld((u16)0);
    }
  }

  float mrow[8], lrow[8];
  f32x8 o0 = {}, o1 = {};
#pragma unroll
  for (int v = 0; v < 8; ++v) { mrow[v] = -1e30f; lrow[v] = 0.0f; }

  for (int ch = 0; ch < 11; ++ch) {
    __syncthreads();                                 // protect prev-iter LDS reads
#if HAVE_ASYNC_LDS
    {
      const int t = threadIdx.x;
      const bool isK = t < 128;
      const int  s = t & 127;                        // one b128 (8 bf16) per lane
      const u16* gsrc = (isK ? kb : vb) + base + (long)ch * 1024 + s * 8;
      u16*       ldst = (isK ? Ks : Vs) + s * 8;
      async_ld_b128(gsrc, ldst);                     // LDS <- global, no VGPR data
      wait_async0();
      if (ch == 10) {                                // zero tail tokens 343..351
        for (int e = 736 + t; e < 1024; e += 256) { Ks[e] = 0; Vs[e] = 0; }
      }
    }
#else
#pragma unroll
    for (int j = 0; j < 4; ++j) {                    // sync fallback staging
      int e = threadIdx.x * 4 + j;
      int tok = e >> 5, d = e & 31;
      int tg = ch * 32 + tok;
      u16 kv = 0, vv = 0;
      if (tg < N_TOK) {
        kv = kb[base + (long)tg * HD + d];
        vv = vb[base + (long)tg * HD + d];
      }
      Ks[e] = kv;
      Vs[e] = vv;
    }
#endif
    __syncthreads();

    // S = q @ K^T  (two 16-col tiles of the 32-token chunk)
    bf16x16 kf0 = *(const bf16x16*)&Ks[lo * 32 + hi * 16];
    bf16x16 kf1 = *(const bf16x16*)&Ks[(lo + 16) * 32 + hi * 16];
    f32x8 z = {};
    f32x8 s0 = __builtin_amdgcn_wmma_f32_16x16x32_bf16(false, qa, false, kf0, (short)0, z, false, false);
    f32x8 s1 = __builtin_amdgcn_wmma_f32_16x16x32_bf16(false, qa, false, kf1, (short)0, z, false, false);

    // bias + mask + online softmax (rows live in 16-lane halves of the wave)
#pragma unroll
    for (int v = 0; v < 8; ++v) {
      int ml = hi * 8 + v;
      int mg = rowBase + ml;
      int c0 = ch * 32 + lo, c1 = c0 + 16;
      float sv0 = s0[v], sv1 = s1[v];
      if (mg < N_TOK && c0 < N_TOK) {
        int ri = relidx[mg * N_TOK + c0];
        sv0 += rpb[ri * NH + h] + mask[((long)wmsk * N_TOK + mg) * N_TOK + c0];
      } else sv0 = -1e30f;
      if (mg < N_TOK && c1 < N_TOK) {
        int ri = relidx[mg * N_TOK + c1];
        sv1 += rpb[ri * NH + h] + mask[((long)wmsk * N_TOK + mg) * N_TOK + c1];
      } else sv1 = -1e30f;

      float rm = fmaxf(sv0, sv1);
#pragma unroll
      for (int x = 1; x < 16; x <<= 1) rm = fmaxf(rm, __shfl_xor(rm, x, 32));
      float mn = fmaxf(mrow[v], rm);
      float corr = __expf(mrow[v] - mn);
      mrow[v] = mn;
      sv0 = __expf(sv0 - mn);
      sv1 = __expf(sv1 - mn);
      float rs = sv0 + sv1;
#pragma unroll
      for (int x = 1; x < 16; x <<= 1) rs += __shfl_xor(rs, x, 32);
      lrow[v] = lrow[v] * corr + rs;
      o0[v] *= corr; o1[v] *= corr;
      Pst[wave * 512 + ml * 32 + lo]      = f2bf(sv0);   // C-layout -> LDS
      Pst[wave * 512 + ml * 32 + 16 + lo] = f2bf(sv1);
    }
    __syncthreads();

    // P (A-layout from LDS) @ V chunk -> accumulate O
    bf16x16 pa;
#pragma unroll
    for (int i = 0; i < 8; ++i) {
      pa[i]     = bfld(Pst[wave * 512 + lo * 32 + hi * 8 + i]);
      pa[i + 8] = bfld(Pst[wave * 512 + lo * 32 + 16 + hi * 8 + i]);
    }
    bf16x16 vf0, vf1;
#pragma unroll
    for (int i = 0; i < 16; ++i) {                   // B[k=token][n=d], strided
      int kk = (i & 8) * 2 + hi * 8 + (i & 7);
      vf0[i] = bfld(Vs[kk * 32 + lo]);
      vf1[i] = bfld(Vs[kk * 32 + 16 + lo]);
    }
    o0 = __builtin_amdgcn_wmma_f32_16x16x32_bf16(false, pa, false, vf0, (short)0, o0, false, false);
    o1 = __builtin_amdgcn_wmma_f32_16x16x32_bf16(false, pa, false, vf1, (short)0, o1, false, false);
  }

  // normalize + store (B, N, C) bf16 for the projection GEMM
#pragma unroll
  for (int v = 0; v < 8; ++v) {
    int mg = rowBase + hi * 8 + v;
    if (mg < N_TOK) {
      float inv = 1.0f / lrow[v];
      long o = ((long)b * N_TOK + mg) * CDIM + h * HD;
      aob[o + lo]      = f2bf(o0[v] * inv);
      aob[o + 16 + lo] = f2bf(o1[v] * inv);
    }
  }
}

// ---------------------------------------------------------------- output projection
__global__ void __launch_bounds__(256) proj_gemm(
    const u16* __restrict__ ab, const u16* __restrict__ wb,
    const float* __restrict__ pb, float* __restrict__ out) {
  const int wave = threadIdx.x >> 5, lane = threadIdx.x & 31;
  const int lo = lane & 15, hi = lane >> 4;
  const int tile = blockIdx.x * 8 + wave;           // 32928 tiles
  const int tm = tile / 12, tn = tile % 12;
  const u16* arow = ab + (long)(tm * 16 + lo) * CDIM;
  const u16* brow = wb + (long)(tn * 16 + lo) * CDIM;
  __builtin_prefetch(brow, 0, 3);
  f32x8 acc = {};
#pragma unroll
  for (int ks = 0; ks < 6; ++ks) {
    const int k0 = ks * 32;
    bf16x16 a, b;
#pragma unroll
    for (int i = 0; i < 8; ++i) {
      a[i]     = bfld(arow[k0 + hi * 8 + i]);
      a[i + 8] = bfld(arow[k0 + 16 + hi * 8 + i]);
    }
#pragma unroll
    for (int i = 0; i < 16; ++i) b[i] = bfld(brow[k0 + (i & 8) * 2 + hi * 8 + (i & 7)]);
    acc = __builtin_amdgcn_wmma_f32_16x16x32_bf16(false, a, false, b,
                                                  (short)0, acc, false, false);
  }
  const int jc = tn * 16 + lo;
  const float bj = pb[jc];
#pragma unroll
  for (int v = 0; v < 8; ++v)
    out[(long)(tm * 16 + hi * 8 + v) * CDIM + jc] = acc[v] + bj;
}

// ---------------------------------------------------------------- launcher
extern "C" void kernel_launch(void* const* d_in, const int* in_sizes, int n_in,
                              void* d_out, int out_size, void* d_ws, size_t ws_size,
                              hipStream_t stream) {
  const float* x      = (const float*)d_in[0];
  const float* mask   = (const float*)d_in[1];
  const float* qkv_w  = (const float*)d_in[2];
  const float* qkv_b  = (const float*)d_in[3];
  const float* proj_w = (const float*)d_in[4];
  const float* proj_b = (const float*)d_in[5];
  const float* rpb    = (const float*)d_in[6];
  const int*   reli   = (const int*)d_in[7];
  float* out = (float*)d_out;

  char* ws = (char*)d_ws;                 // all offsets 256B-aligned
  u16* xb    = (u16*)(ws + 0);            // 43904*192 bf16
  u16* qkvwb = (u16*)(ws + 16859136);     // 576*192
  u16* prjwb = (u16*)(ws + 17080320);     // 192*192
  u16* qbuf  = (u16*)(ws + 17154048);     // (B,H,N,32)
  u16* kbuf  = (u16*)(ws + 34013184);
  u16* vbuf  = (u16*)(ws + 50872320);
  u16* aob   = (u16*)(ws + 67731456);     // attention output, bf16

  cvt_bf16<<<8429568 / 256, 256, 0, stream>>>(x, xb, 8429568);
  cvt_bf16<<<110592 / 256, 256, 0, stream>>>(qkv_w, qkvwb, 110592);
  cvt_bf16<<<36864 / 256, 256, 0, stream>>>(proj_w, prjwb, 36864);
  qkv_gemm<<<12348, 256, 0, stream>>>(xb, qkvwb, qkv_b, qbuf, kbuf, vbuf);
  win_attn<<<dim3(768, 3), 256, 0, stream>>>(qbuf, kbuf, vbuf, mask, rpb, reli, aob);
  proj_gemm<<<4116, 256, 0, stream>>>(aob, prjwb, proj_b, out);
}
